// JAXAttention_47725676593806
// MI455X (gfx1250) — matrix operand order repaired and checked by reading, not compiled, not run on hardware
//
#include <hip/hip_runtime.h>
#include <hip/hip_bf16.h>

#define BATCH 8
#define NSEQ  2048
#define DHID  32

typedef _Float16 v16h __attribute__((ext_vector_type(16)));
typedef _Float16 v8h  __attribute__((ext_vector_type(8)));
typedef float    v8f  __attribute__((ext_vector_type(8)));

// ---------------------------------------------------------------------------
// Kernel 1: q = s @ Qw, k = s @ Kw  (fp32 compute, f16 store into workspace)
// One thread per (b, n) row. Tiny: 16384 threads, ~320 FMAs each.
// ---------------------------------------------------------------------------
__global__ __launch_bounds__(256) void qk_project_kernel(
    const float* __restrict__ s, const float* __restrict__ Qw,
    const float* __restrict__ Kw, _Float16* __restrict__ qh,
    _Float16* __restrict__ kh) {
  __shared__ float wq[5 * DHID];
  __shared__ float wk[5 * DHID];
  if (threadIdx.x < 5 * DHID) {
    wq[threadIdx.x] = Qw[threadIdx.x];
    wk[threadIdx.x] = Kw[threadIdx.x];
  }
  __syncthreads();

  const int idx = blockIdx.x * 256 + threadIdx.x;
  if (idx >= BATCH * NSEQ) return;

  float sv[5];
#pragma unroll
  for (int d = 0; d < 5; ++d) sv[d] = s[(size_t)idx * 5 + d];

#pragma unroll
  for (int m = 0; m < DHID; ++m) {
    float q = 0.f, k = 0.f;
#pragma unroll
    for (int d = 0; d < 5; ++d) {
      q = fmaf(sv[d], wq[d * DHID + m], q);
      k = fmaf(sv[d], wk[d * DHID + m], k);
    }
    qh[(size_t)idx * DHID + m] = (_Float16)q;
    kh[(size_t)idx * DHID + m] = (_Float16)k;
  }
}

// ---------------------------------------------------------------------------
// Kernel 2: att = ((q k^T)^2 * G) row-normalized.
// grid = (N/16, B); block = 256 (8 waves). Wave w owns column tiles
// j0 = (w*16 + jt)*16, jt = 0..15. Numerator tiles kept in VGPRs across the
// normalization, so G is read once and out written once (HBM-optimal).
// ---------------------------------------------------------------------------
__global__ __launch_bounds__(256) void att_wmma_kernel(
    const float* __restrict__ G, const _Float16* __restrict__ qh,
    const _Float16* __restrict__ kh, float* __restrict__ out,
    unsigned long long gbatch_stride) {
  const int wave = threadIdx.x >> 5;
  const int lane = threadIdx.x & 31;
  const int n    = lane & 15;   // column within 16x16 tile / row M for A loads
  const int hs   = lane >> 4;   // half-select
  const int i0   = blockIdx.x * 16;
  const int b    = blockIdx.y;

  // ---- Load A (16x32 f16 q-tile). Contiguous layout matching B: lanes 0-15
  // hold K=0..15, lanes 16-31 hold K=16..31.
  const _Float16* qrow = qh + (size_t)(b * NSEQ + i0 + n) * DHID;
  v8h alo = *(const v8h*)(qrow + 16 * hs);       // K = 16*hs .. 16*hs+7
  v8h ahi = *(const v8h*)(qrow + 16 * hs + 8);   // K = 16*hs+8 .. +7
  v16h a;
#pragma unroll
  for (int t = 0; t < 8; ++t) { a[t] = alo[t]; a[8 + t] = ahi[t]; }

  const float* gbase = G + (size_t)b * gbatch_stride + (size_t)i0 * NSEQ;
  float*       obase = out + ((size_t)b * NSEQ + i0) * (size_t)NSEQ;

  v8f   acc[16];
  float psum[8];
#pragma unroll
  for (int d = 0; d < 8; ++d) psum[d] = 0.f;

#pragma unroll
  for (int jt = 0; jt < 16; ++jt) {
    const int j0 = (wave * 16 + jt) * 16;

    // ---- Load B (k^T tile, 32x16 f16). Same contiguous k order as A:
    // lane = column n, lanes 0-15 hold K=0..15 contiguously,
    // lanes 16-31 hold K=16..31. Column n of B == row j0+n of k.
    const _Float16* krow = kh + (size_t)(b * NSEQ + j0 + n) * DHID;
    v16h bv = *(const v16h*)(krow + 16 * hs);

    v8f c = {};
    c = __builtin_amdgcn_wmma_f32_16x16x32_f16(
        /*neg_a=*/false, a, /*neg_b=*/false, bv,
        /*c_mod=*/(short)0, c, /*reuse_a=*/false, /*reuse_b=*/false);

    // C/D layout: VGPR d -> row M = d + 8*hs, column N = lane&15.
    v8f r;
#pragma unroll
    for (int d = 0; d < 8; ++d) {
      const float g = gbase[(size_t)(d + 8 * hs) * NSEQ + j0 + n];
      const float num = c[d] * c[d] * g;
      r[d] = num;
      psum[d] += num;
    }
    acc[jt] = r;
  }

  // ---- Row-sum reduction across the 16 lanes sharing each row half.
#pragma unroll
  for (int d = 0; d < 8; ++d) {
    float v = psum[d];
#pragma unroll
    for (int mask = 1; mask < 16; mask <<= 1) v += __shfl_xor(v, mask, 32);
    psum[d] = v;  // full sum of this wave's 16 tiles for row d + 8*hs
  }

  __shared__ float partial[8][16];  // [wave][row]
  if (n == 0) {
#pragma unroll
    for (int d = 0; d < 8; ++d) partial[wave][d + 8 * hs] = psum[d];
  }
  __syncthreads();

  float rinv[8];
#pragma unroll
  for (int d = 0; d < 8; ++d) {
    const int row = d + 8 * hs;
    float sum = 0.001f;
#pragma unroll
    for (int wv = 0; wv < 8; ++wv) sum += partial[wv][row];
    rinv[d] = 1.0f / sum;
  }

  // ---- Scale register-resident numerators and write output (coalesced:
  // lanes 0-15 cover 64B contiguous per row).
#pragma unroll
  for (int jt = 0; jt < 16; ++jt) {
    const int j0 = (wave * 16 + jt) * 16;
#pragma unroll
    for (int d = 0; d < 8; ++d) {
      obase[(size_t)(d + 8 * hs) * NSEQ + j0 + n] = acc[jt][d] * rinv[d];
    }
  }
}

// ---------------------------------------------------------------------------
extern "C" void kernel_launch(void* const* d_in, const int* in_sizes, int n_in,
                              void* d_out, int out_size, void* d_ws,
                              size_t ws_size, hipStream_t stream) {
  const float* s  = (const float*)d_in[0];
  const float* G  = (const float*)d_in[1];
  const float* Qw = (const float*)d_in[2];
  const float* Kw = (const float*)d_in[3];
  float* out = (float*)d_out;

  _Float16* qh = (_Float16*)d_ws;                       // B*N*32 f16 = 1 MB
  _Float16* kh = qh + (size_t)BATCH * NSEQ * DHID;      // + 1 MB

  qk_project_kernel<<<(BATCH * NSEQ + 255) / 256, 256, 0, stream>>>(
      s, Qw, Kw, qh, kh);

  // Gmat may be [B,N,N] or broadcast [N,N].
  const unsigned long long gstride =
      (in_sizes[1] == NSEQ * NSEQ) ? 0ull
                                   : (unsigned long long)NSEQ * NSEQ;

  dim3 grid(NSEQ / 16, BATCH);
  att_wmma_kernel<<<grid, 256, 0, stream>>>(G, qh, kh, out, gstride);
}
